// WindFarmModel_12902081757909
// MI455X (gfx1250) — matrix operand order repaired and checked by reading, not compiled, not run on hardware
//
#include <hip/hip_runtime.h>
#include <hip/hip_bf16.h>

typedef __attribute__((ext_vector_type(16))) _Float16 v16h;
typedef __attribute__((ext_vector_type(8)))  float    v8f;

namespace cfg {
constexpr int B = 4, T = 24, N = 512, BN = 2048, ROWS = T * BN, C = 64;
constexpr int NE_STATIC = 4096;
constexpr int NE_STAT = B * NE_STATIC;          // 16384
constexpr int KNN = 5;
constexpr int NE_DYN = B * N * KNN;             // 10240
constexpr int NLOOP = BN;                       // 2048
constexpr int NE_TOT = NE_STAT + NE_DYN + NLOOP; // 28672
}
using namespace cfg;

// ---------------------------------------------------------------- WMMA helpers
__device__ __forceinline__ v8f wmma_f16(v16h a, v16h b, v8f c) {
  return __builtin_amdgcn_wmma_f32_16x16x32_f16(false, a, false, b, (short)0, c,
                                                false, false);
}

// A fragment: 16x32 f16 from f32 row-major A (lda = K)
__device__ __forceinline__ v16h load_a_frag(const float* A, int lda, int m0,
                                            int k0, int lane) {
  v16h a;
  const float* p = A + (size_t)(m0 + (lane & 15)) * lda + k0 + ((lane >> 4) << 3);
#pragma unroll
  for (int h = 0; h < 16; ++h)
    a[h] = (_Float16)p[((h >> 3) << 4) + (h & 7)];
  return a;
}

// B fragment: 32x16 f16 from f32 row-major B (ldb = N)
__device__ __forceinline__ v16h load_b_frag(const float* Bm, int ldb, int k0,
                                            int n0, int lane) {
  v16h b;
  const float* p = Bm + (size_t)(k0 + ((lane >> 4) << 3)) * ldb + n0 + (lane & 15);
#pragma unroll
  for (int h = 0; h < 16; ++h)
    b[h] = (_Float16)p[(size_t)(((h >> 3) << 4) + (h & 7)) * ldb];
  return b;
}

__device__ __forceinline__ void atomicMaxF(float* addr, float val) {
  int* ia = (int*)addr;
  int old = __float_as_int(*addr);
  while (__int_as_float(old) < val) {
    int assumed = old;
    old = atomicCAS(ia, assumed, __float_as_int(val));
    if (old == assumed) break;
  }
}

// ---------------------------------------------------------------- generic GEMM
// Out[M,N] = act( A[M,K] @ B[K,N] + bias + resid ), one wave per 16x16 tile
__global__ void gemm_wmma(const float* __restrict__ A, const float* __restrict__ Bm,
                          const float* __restrict__ bias,
                          const float* __restrict__ resid, float* __restrict__ Out,
                          int M, int K, int Nn, int relu_flag) {
  int lane = threadIdx.x & 31;
  int ntn = Nn >> 4;
  int m0 = (blockIdx.x / ntn) << 4;
  int n0 = (blockIdx.x % ntn) << 4;
  v8f c = {};
  for (int k0 = 0; k0 < K; k0 += 32) {
    v16h a = load_a_frag(A, K, m0, k0, lane);
    v16h b = load_b_frag(Bm, Nn, k0, n0, lane);
    c = wmma_f16(a, b, c);
  }
  int col = n0 + (lane & 15);
  float bb = bias ? bias[col] : 0.f;
#pragma unroll
  for (int j = 0; j < 8; ++j) {
    int row = m0 + j + ((lane >> 4) << 3);
    float v = c[j] + bb;
    if (resid) v += resid[(size_t)row * Nn + col];
    if (relu_flag) v = fmaxf(v, 0.f);
    Out[(size_t)row * Nn + col] = v;
  }
}

// ---------------------------------------------------------------- fused FFN
// Out = X + relu(X@W1 + b1)@W2 + b2   (X:[M,64], W1:[64,2048], W2:[2048,64])
__global__ void ffn_fused(const float* __restrict__ X, const float* __restrict__ W1,
                          const float* __restrict__ b1, const float* __restrict__ W2,
                          const float* __restrict__ b2, float* __restrict__ Out) {
  __shared__ _Float16 lds[16 * 32];
  int lane = threadIdx.x & 31;
  int m0 = blockIdx.x << 4;
  v16h a0 = load_a_frag(X, 64, m0, 0, lane);
  v16h a1 = load_a_frag(X, 64, m0, 32, lane);
  v8f acc[4] = {v8f{}, v8f{}, v8f{}, v8f{}};
  int colL = lane & 15;
  int r0 = (lane >> 4) << 3;
  for (int hc = 0; hc < 2048; hc += 32) {
    __builtin_prefetch(W1 + hc + 32, 0, 1);
    __builtin_prefetch(W2 + (size_t)(hc + 32) * 64, 0, 1);
    // hidden = relu(X @ W1[:, hc:hc+32] + b1) : two 16-col subtiles
#pragma unroll
    for (int j = 0; j < 2; ++j) {
      v8f d = {};
      d = wmma_f16(a0, load_b_frag(W1, 2048, 0, hc + j * 16, lane), d);
      d = wmma_f16(a1, load_b_frag(W1, 2048, 32, hc + j * 16, lane), d);
      float bb = b1[hc + j * 16 + colL];
#pragma unroll
      for (int i = 0; i < 8; ++i)
        lds[(r0 + i) * 32 + j * 16 + colL] = (_Float16)fmaxf(d[i] + bb, 0.f);
    }
    __syncthreads();
    // reload hidden tile as A fragment (local K = 0..31)
    v16h a2;
    {
      int row = lane & 15;
      int kb = (lane >> 4) << 3;
#pragma unroll
      for (int h = 0; h < 16; ++h)
        a2[h] = lds[row * 32 + kb + ((h >> 3) << 4) + (h & 7)];
    }
    __syncthreads();
#pragma unroll
    for (int j2 = 0; j2 < 4; ++j2)
      acc[j2] = wmma_f16(a2, load_b_frag(W2, 64, hc, j2 * 16, lane), acc[j2]);
  }
#pragma unroll
  for (int j2 = 0; j2 < 4; ++j2) {
    int col = j2 * 16 + colL;
    float bb = b2[col];
#pragma unroll
    for (int i = 0; i < 8; ++i) {
      int row = m0 + i + r0;
      Out[(size_t)row * 64 + col] = acc[j2][i] + bb + X[(size_t)row * 64 + col];
    }
  }
}

// ---------------------------------------------------------------- local encoder
__global__ void local_enc(const float* __restrict__ xloc, const float* __restrict__ w1,
                          const float* __restrict__ b1, const float* __restrict__ g1,
                          const float* __restrict__ e1, const float* __restrict__ w2,
                          const float* __restrict__ b2, const float* __restrict__ g2,
                          const float* __restrict__ e2, float* __restrict__ zflat) {
  int img = blockIdx.x * blockDim.x + threadIdx.x;
  if (img >= ROWS) return;
  float in[63];
  const float* ip = xloc + (size_t)img * 63;
#pragma unroll
  for (int c = 0; c < 63; ++c) in[c] = ip[c];
  float a1[72];
#pragma unroll
  for (int o = 0; o < 8; ++o)
    for (int y = 0; y < 3; ++y)
      for (int x = 0; x < 3; ++x) {
        float acc = b1[o];
        for (int i = 0; i < 7; ++i)
          for (int ky = 0; ky < 3; ++ky)
            for (int kx = 0; kx < 3; ++kx) {
              int yy = y + ky - 1, xx = x + kx - 1;
              if (yy >= 0 && yy < 3 && xx >= 0 && xx < 3)
                acc += w1[((o * 7 + i) * 3 + ky) * 3 + kx] * in[(i * 3 + yy) * 3 + xx];
            }
        a1[(o * 3 + y) * 3 + x] = acc;
      }
#pragma unroll
  for (int gg = 0; gg < 2; ++gg) {
    float m = 0.f, v = 0.f;
    for (int c = gg * 4; c < gg * 4 + 4; ++c)
      for (int p = 0; p < 9; ++p) m += a1[c * 9 + p];
    m *= (1.f / 36.f);
    for (int c = gg * 4; c < gg * 4 + 4; ++c)
      for (int p = 0; p < 9; ++p) { float d = a1[c * 9 + p] - m; v += d * d; }
    v *= (1.f / 36.f);
    float inv = rsqrtf(v + 1e-5f);
    for (int c = gg * 4; c < gg * 4 + 4; ++c)
      for (int p = 0; p < 9; ++p) {
        float x = (a1[c * 9 + p] - m) * inv * g1[c] + e1[c];
        a1[c * 9 + p] = x / (1.f + expf(-x));
      }
  }
  float a2[32];
#pragma unroll
  for (int o = 0; o < 8; ++o)
    for (int y = 0; y < 2; ++y)
      for (int x = 0; x < 2; ++x) {
        float acc = b2[o];
        for (int i = 0; i < 8; ++i)
          for (int ky = 0; ky < 3; ++ky)
            for (int kx = 0; kx < 3; ++kx) {
              int yy = 2 * y + ky - 1, xx = 2 * x + kx - 1;
              if (yy >= 0 && yy < 3 && xx >= 0 && xx < 3)
                acc += w2[((o * 8 + i) * 3 + ky) * 3 + kx] * a1[(i * 3 + yy) * 3 + xx];
            }
        a2[(o * 2 + y) * 2 + x] = acc;
      }
#pragma unroll
  for (int gg = 0; gg < 2; ++gg) {
    float m = 0.f, v = 0.f;
    for (int c = gg * 4; c < gg * 4 + 4; ++c)
      for (int p = 0; p < 4; ++p) m += a2[c * 4 + p];
    m *= (1.f / 16.f);
    for (int c = gg * 4; c < gg * 4 + 4; ++c)
      for (int p = 0; p < 4; ++p) { float d = a2[c * 4 + p] - m; v += d * d; }
    v *= (1.f / 16.f);
    float inv = rsqrtf(v + 1e-5f);
    for (int c = gg * 4; c < gg * 4 + 4; ++c)
      for (int p = 0; p < 4; ++p) {
        float x = (a2[c * 4 + p] - m) * inv * g2[c] + e2[c];
        a2[c * 4 + p] = x / (1.f + expf(-x));
      }
  }
  float* zp = zflat + (size_t)img * 32;
#pragma unroll
  for (int c = 0; c < 32; ++c) zp[c] = a2[c];
}

// ---------------------------------------------------------------- node features
// X[t,b,n,c]:  c<32 -> x_base[b,t,n,:]@st_w + st_b ; c>=32 -> zflat scramble
__global__ void node_feat(const float* __restrict__ x_base, const float* __restrict__ st_w,
                          const float* __restrict__ st_b, const float* __restrict__ zflat,
                          float* __restrict__ X) {
  int g = blockIdx.x * blockDim.x + threadIdx.x;
  if (g >= ROWS * C) return;
  int row = g >> 6, c = g & 63;
  int t = row / BN, bn = row % BN, b = bn >> 9, n = bn & 511;
  float v;
  if (c < 32) {
    const float* xb = x_base + ((size_t)(b * T + t) * N + n) * 10;
    v = st_b[c];
#pragma unroll
    for (int r = 0; r < 10; ++r) v += xb[r] * st_w[r * 32 + c];
  } else {
    v = zflat[((size_t)b * T * N + (size_t)n * T + t) * 32 + (c - 32)];
  }
  X[g] = v;
}

__global__ void row_norm(const float* __restrict__ X, float* __restrict__ rn) {
  int row = blockIdx.x * blockDim.x + threadIdx.x;
  if (row >= ROWS) return;
  const float* p = X + (size_t)row * C;
  float s = 0.f;
#pragma unroll
  for (int c = 0; c < C; ++c) s += p[c] * p[c];
  rn[row] = fmaxf(sqrtf(s), 1e-8f);
}

// ---------------------------------------------------------------- kNN (top-5)
__global__ void topk_knn(const float* __restrict__ X, const float* __restrict__ rn,
                         int* __restrict__ dyn) {
  int row = blockIdx.x * blockDim.x + threadIdx.x;
  if (row >= ROWS) return;
  int t = row / BN, bn = row % BN, b = bn >> 9, i = bn & 511;
  float xi[64];
  const float* xip = X + (size_t)row * C;
#pragma unroll
  for (int c = 0; c < C; ++c) xi[c] = xip[c];
  float ni = rn[row];
  float vb[KNN];
  int ib[KNN];
#pragma unroll
  for (int k = 0; k < KNN; ++k) { vb[k] = -__builtin_inff(); ib[k] = 0; }
  int base = t * BN + b * N;
  for (int j = 0; j < N; ++j) {
    const float* xj = X + (size_t)(base + j) * C;
    float dot = 0.f;
#pragma unroll
    for (int c = 0; c < C; ++c) dot += xi[c] * xj[c];
    float cs = dot / (ni * rn[base + j]);
    if (cs > vb[KNN - 1]) {                  // strict > keeps lax.top_k tie order
      int p = KNN - 1;
      while (p > 0 && cs > vb[p - 1]) { vb[p] = vb[p - 1]; ib[p] = ib[p - 1]; --p; }
      vb[p] = cs; ib[p] = j;
    }
  }
  int ob = t * NE_DYN + b * (N * KNN) + i * KNN;
#pragma unroll
  for (int k = 0; k < KNN; ++k) dyn[ob + k] = ib[k];
}

// ---------------------------------------------------------------- GATv2 edges
__device__ __forceinline__ void edge_decode(int e, int t, const int* __restrict__ ei,
                                            const int* __restrict__ dyn, int& src,
                                            int& dst, bool& valid) {
  if (e < NE_STAT) {
    int b = e >> 12, j = e & 4095;
    src = ei[j] + b * N;
    dst = ei[NE_STATIC + j] + b * N;
    valid = (src != dst);
  } else if (e < NE_STAT + NE_DYN) {
    int d = e - NE_STAT;
    src = (d % (N * KNN)) / KNN;             // faithful: no batch offset
    dst = dyn[t * NE_DYN + d];
    valid = (src != dst);
  } else {
    src = dst = e - (NE_STAT + NE_DYN);
    valid = true;
  }
}

__global__ void fill_f32(float* p, float v, int n) {
  int g = blockIdx.x * blockDim.x + threadIdx.x;
  if (g < n) p[g] = v;
}
__global__ void bias_rows(float* __restrict__ p, const float* __restrict__ bias, int rows) {
  int g = blockIdx.x * blockDim.x + threadIdx.x;
  if (g < rows * C) p[g] = bias[g & 63];
}
__global__ void relu_ip(float* p, int n) {
  int g = blockIdx.x * blockDim.x + threadIdx.x;
  if (g < n) p[g] = fmaxf(p[g], 0.f);
}

__global__ void edge_score(const float* __restrict__ XL, const float* __restrict__ XR,
                           const float* __restrict__ att, const int* __restrict__ ei,
                           const int* __restrict__ dyn, float* __restrict__ ebuf,
                           float* __restrict__ emax) {
  int g = blockIdx.x * blockDim.x + threadIdx.x;
  if (g >= T * NE_TOT) return;
  int t = g / NE_TOT, e = g % NE_TOT;
  int src, dst; bool valid;
  edge_decode(e, t, ei, dyn, src, dst, valid);
  const float* xl = XL + (size_t)(t * BN + src) * C;
  const float* xr = XR + (size_t)(t * BN + dst) * C;
  float s = 0.f;
#pragma unroll
  for (int c = 0; c < C; ++c) {
    float v = xl[c] + xr[c];
    v = (v > 0.f) ? v : 0.2f * v;
    s += v * att[c];
  }
  ebuf[g] = s;
  if (valid) atomicMaxF(&emax[t * BN + dst], s);
}

__global__ void edge_num(const float* __restrict__ ebuf, const float* __restrict__ emax,
                         const int* __restrict__ ei, const int* __restrict__ dyn,
                         float* __restrict__ numbuf, float* __restrict__ den) {
  int g = blockIdx.x * blockDim.x + threadIdx.x;
  if (g >= T * NE_TOT) return;
  int t = g / NE_TOT, e = g % NE_TOT;
  int src, dst; bool valid;
  edge_decode(e, t, ei, dyn, src, dst, valid);
  float nv = valid ? expf(ebuf[g] - emax[t * BN + dst]) : 0.f;
  numbuf[g] = nv;
  if (valid) atomicAdd(&den[t * BN + dst], nv);
}

__global__ void edge_agg(const float* __restrict__ numbuf, const float* __restrict__ den,
                         const float* __restrict__ XL, const int* __restrict__ ei,
                         const int* __restrict__ dyn, float* __restrict__ H) {
  int g = blockIdx.x * blockDim.x + threadIdx.x;
  if (g >= T * NE_TOT * C) return;
  int c = g & 63;
  int e6 = g >> 6;
  int t = e6 / NE_TOT, e = e6 % NE_TOT;
  int src, dst; bool valid;
  edge_decode(e, t, ei, dyn, src, dst, valid);
  float nv = numbuf[t * NE_TOT + e];
  if (!valid || nv == 0.f) return;
  float alpha = nv / (den[t * BN + dst] + 1e-16f);
  atomicAdd(&H[(size_t)(t * BN + dst) * C + c],
            alpha * XL[(size_t)(t * BN + src) * C + c]);
}

// ---------------------------------------------------------------- temporal conv
__global__ void tconvA(const float* __restrict__ H, const float* __restrict__ w1,
                       const float* __restrict__ b1, float* __restrict__ Ct) {
  int g = blockIdx.x * blockDim.x + threadIdx.x;
  if (g >= ROWS * C) return;
  int row = g >> 6, o = g & 63;
  int s = row >> 9, n = row & 511;
  float acc = b1[o];
#pragma unroll
  for (int k = 0; k < 3; ++k) {
    int nn = n + k - 1;
    if (nn < 0 || nn >= N) continue;
    const float* hp = H + (size_t)(s * N + nn) * C;
    for (int i = 0; i < C; ++i) acc += w1[(o * C + i) * 3 + k] * hp[i];
  }
  Ct[g] = fmaxf(acc, 0.f);
}

__global__ void tconvB(const float* __restrict__ Ct, const float* __restrict__ w2,
                       const float* __restrict__ b2, float* __restrict__ H) {
  int g = blockIdx.x * blockDim.x + threadIdx.x;
  if (g >= ROWS * C) return;
  int row = g >> 6, o = g & 63;
  int s = row >> 9, n = row & 511;
  float acc = b2[o];
#pragma unroll
  for (int k = 0; k < 3; ++k) {
    int nn = n + 2 * (k - 1);
    if (nn < 0 || nn >= N) continue;
    const float* cp = Ct + (size_t)(s * N + nn) * C;
    for (int i = 0; i < C; ++i) acc += w2[(o * C + i) * 3 + k] * cp[i];
  }
  int t = s / B;                              // s = t*B + b
  if (t > 0) H[g] += acc;
}

// (t,b) -> (b,t) row remap
__global__ void remap_bt(const float* __restrict__ H, float* __restrict__ X2) {
  int g = blockIdx.x * blockDim.x + threadIdx.x;
  if (g >= ROWS * C) return;
  int row2 = g >> 6, c = g & 63;
  int bt = row2 >> 9, n = row2 & 511;
  int b = bt / T, t = bt % T;
  X2[g] = H[((size_t)(t * B + b) * N + n) * C + c];
}

// ---------------------------------------------------------------- attention
__global__ void attn_kernel(const float* __restrict__ QKV, float* __restrict__ O) {
  __shared__ float Ks[96 * 16];
  __shared__ float Vs[96 * 16];
  int n = blockIdx.x, h = blockIdx.y, i = threadIdx.x;   // i in [0,96)
  const float* kp = QKV + ((size_t)i * N + n) * 192 + 64 + h * 16;
  const float* vp = QKV + ((size_t)i * N + n) * 192 + 128 + h * 16;
#pragma unroll
  for (int d = 0; d < 16; ++d) { Ks[i * 16 + d] = kp[d]; Vs[i * 16 + d] = vp[d]; }
  __syncthreads();
  float q[16];
  const float* qp = QKV + ((size_t)i * N + n) * 192 + h * 16;
#pragma unroll
  for (int d = 0; d < 16; ++d) q[d] = qp[d];
  float m = -__builtin_inff();
  for (int j = 0; j < 96; ++j) {
    float s = 0.f;
#pragma unroll
    for (int d = 0; d < 16; ++d) s += q[d] * Ks[j * 16 + d];
    m = fmaxf(m, s * 0.25f);
  }
  float l = 0.f, o[16] = {};
  for (int j = 0; j < 96; ++j) {
    float s = 0.f;
#pragma unroll
    for (int d = 0; d < 16; ++d) s += q[d] * Ks[j * 16 + d];
    float ex = expf(s * 0.25f - m);
    l += ex;
#pragma unroll
    for (int d = 0; d < 16; ++d) o[d] += ex * Vs[j * 16 + d];
  }
  float* op = O + ((size_t)i * N + n) * C + h * 16;
  float rl = 1.f / l;
#pragma unroll
  for (int d = 0; d < 16; ++d) op[d] = o[d] * rl;
}

// ---------------------------------------------------------------- layernorm
__global__ void ln_kernel(const float* __restrict__ Xin, const float* __restrict__ g,
                          const float* __restrict__ bb, float* __restrict__ Xout) {
  int row = blockIdx.x, lane = threadIdx.x;  // 32 lanes, 2 elems each
  const float* p = Xin + (size_t)row * C;
  float v0 = p[lane], v1 = p[lane + 32];
  float s = v0 + v1, sq = v0 * v0 + v1 * v1;
#pragma unroll
  for (int o = 16; o > 0; o >>= 1) {
    s += __shfl_xor(s, o, 32);
    sq += __shfl_xor(sq, o, 32);
  }
  float mean = s * (1.f / 64.f);
  float var = sq * (1.f / 64.f) - mean * mean;
  float inv = rsqrtf(var + 1e-5f);
  float* q = Xout + (size_t)row * C;
  q[lane] = (v0 - mean) * inv * g[lane] + bb[lane];
  q[lane + 32] = (v1 - mean) * inv * g[lane + 32] + bb[lane + 32];
}

// ---------------------------------------------------------------- head
__global__ void tmax_kernel(const float* __restrict__ X3, float* __restrict__ tmax) {
  int g = blockIdx.x * blockDim.x + threadIdx.x;
  if (g >= B * N * C) return;
  int b = g / (N * C), r = g % (N * C), n = r >> 6, c = r & 63;
  float m = -__builtin_inff();
  for (int t = 0; t < T; ++t)
    m = fmaxf(m, X3[((size_t)(b * T + t) * N + n) * C + c]);
  tmax[g] = m;
}

__global__ void combo_kernel(const float* __restrict__ fus_w, const float* __restrict__ fus_b,
                             const float* __restrict__ reg_w, const float* __restrict__ reg_b,
                             float* __restrict__ combo) {
  int tid = threadIdx.x;
  if (tid < 64) {
    float s = 0.f;
    for (int c = 0; c < 64; ++c) s += fus_w[(1 + tid) * 64 + c] * reg_w[c];
    combo[tid] = s;
  } else if (tid == 64) {
    float s = 0.f;
    for (int c = 0; c < 64; ++c) s += fus_w[c] * reg_w[c];
    combo[64] = s;
  } else if (tid == 65) {
    float s = reg_b[0];
    for (int c = 0; c < 64; ++c) s += fus_b[c] * reg_w[c];
    combo[65] = s;
  }
}

__global__ void final_kernel(const float* __restrict__ X3, const float* __restrict__ tmax,
                             const float* __restrict__ combo, float* __restrict__ out) {
  int g = blockIdx.x * blockDim.x + threadIdx.x;   // = (b*T+t)*N + n
  if (g >= ROWS) return;
  int b = g / (T * N), n = g & 511;
  const float* p = X3 + (size_t)g * C;
  const float* tm = tmax + (size_t)(b * N + n) * C;
  float mean = 0.f, dv = 0.f;
#pragma unroll
  for (int c = 0; c < C; ++c) { mean += p[c]; dv += tm[c] * combo[c]; }
  mean *= (1.f / 64.f);
  out[g] = combo[65] + combo[64] * mean + dv;
}

// ================================================================ launch
extern "C" void kernel_launch(void* const* d_in, const int* in_sizes, int n_in,
                              void* d_out, int out_size, void* d_ws, size_t ws_size,
                              hipStream_t stream) {
  const float* x_base = (const float*)d_in[0];
  const float* x_local = (const float*)d_in[1];
  const int* edge_index = (const int*)d_in[2];
  int p = 3;
  const float* enc1_w = (const float*)d_in[p++]; const float* enc1_b = (const float*)d_in[p++];
  const float* gn1_g = (const float*)d_in[p++];  const float* gn1_b = (const float*)d_in[p++];
  const float* enc2_w = (const float*)d_in[p++]; const float* enc2_b = (const float*)d_in[p++];
  const float* gn2_g = (const float*)d_in[p++];  const float* gn2_b = (const float*)d_in[p++];
  const float* st_w = (const float*)d_in[p++];   const float* st_b = (const float*)d_in[p++];
  const float* g1_wl = (const float*)d_in[p++];  const float* g1_bl = (const float*)d_in[p++];
  const float* g1_wr = (const float*)d_in[p++];  const float* g1_br = (const float*)d_in[p++];
  const float* g1_att = (const float*)d_in[p++]; const float* g1_bias = (const float*)d_in[p++];
  const float* g2_wl = (const float*)d_in[p++];  const float* g2_bl = (const float*)d_in[p++];
  const float* g2_wr = (const float*)d_in[p++];  const float* g2_br = (const float*)d_in[p++];
  const float* g2_att = (const float*)d_in[p++]; const float* g2_bias = (const float*)d_in[p++];
  const float* tc_w1 = (const float*)d_in[p++];  const float* tc_b1 = (const float*)d_in[p++];
  const float* tc_w2 = (const float*)d_in[p++];  const float* tc_b2 = (const float*)d_in[p++];
  const float* attn_wqkv = (const float*)d_in[p++]; const float* attn_bqkv = (const float*)d_in[p++];
  const float* attn_wo = (const float*)d_in[p++];   const float* attn_bo = (const float*)d_in[p++];
  const float* ln1_g = (const float*)d_in[p++];  const float* ln1_b = (const float*)d_in[p++];
  const float* ff_w1 = (const float*)d_in[p++];  const float* ff_b1 = (const float*)d_in[p++];
  const float* ff_w2 = (const float*)d_in[p++];  const float* ff_b2 = (const float*)d_in[p++];
  const float* ln2_g = (const float*)d_in[p++];  const float* ln2_b = (const float*)d_in[p++];
  const float* fus_w = (const float*)d_in[p++];  const float* fus_b = (const float*)d_in[p++];
  const float* reg_w = (const float*)d_in[p++];  const float* reg_b = (const float*)d_in[p++];

  size_t off = 0;
  char* wsb = (char*)d_ws;
  auto af = [&](size_t n) { float* r = (float*)(wsb + off); off += n * sizeof(float); return r; };
  auto ai = [&](size_t n) { int* r = (int*)(wsb + off); off += n * sizeof(int); return r; };

  float* zflat = af((size_t)ROWS * 32);
  float* Xn = af((size_t)ROWS * C);
  float* rn = af(ROWS);
  int* dyn = ai((size_t)T * NE_DYN);
  float* XL = af((size_t)ROWS * C);
  float* XR = af((size_t)ROWS * C);
  float* ebuf = af((size_t)T * NE_TOT);
  float* numbuf = af((size_t)T * NE_TOT);
  float* emax = af((size_t)T * BN);
  float* den = af((size_t)T * BN);
  float* H1 = af((size_t)ROWS * C);
  float* H2 = af((size_t)ROWS * C);
  float* Ct = af((size_t)ROWS * C);
  float* X2 = af((size_t)ROWS * C);
  float* QKV = af((size_t)ROWS * 192);
  float* Ob = af((size_t)ROWS * C);
  float* Y1 = af((size_t)ROWS * C);
  float* XN = af((size_t)ROWS * C);
  float* Y2 = af((size_t)ROWS * C);
  float* X3 = af((size_t)ROWS * C);
  float* tmx = af((size_t)B * N * C);
  float* combo = af(66);
  (void)ws_size; (void)n_in; (void)in_sizes; (void)out_size;

  const int TPB = 256;
  const int nEC = ROWS * C;

  // 1. local encoder + node features
  local_enc<<<(ROWS + TPB - 1) / TPB, TPB, 0, stream>>>(
      x_local, enc1_w, enc1_b, gn1_g, gn1_b, enc2_w, enc2_b, gn2_g, gn2_b, zflat);
  node_feat<<<(nEC + TPB - 1) / TPB, TPB, 0, stream>>>(x_base, st_w, st_b, zflat, Xn);

  // 2. dynamic kNN graph
  row_norm<<<(ROWS + TPB - 1) / TPB, TPB, 0, stream>>>(Xn, rn);
  topk_knn<<<(ROWS + 127) / 128, 128, 0, stream>>>(Xn, rn, dyn);

  dim3 g64((ROWS / 16) * (C / 16));
  int nE = T * NE_TOT;
  // 3. GATv2 layer 1 (relu)
  gemm_wmma<<<g64, 32, 0, stream>>>(Xn, g1_wl, g1_bl, nullptr, XL, ROWS, C, C, 0);
  gemm_wmma<<<g64, 32, 0, stream>>>(Xn, g1_wr, g1_br, nullptr, XR, ROWS, C, C, 0);
  fill_f32<<<(T * BN + TPB - 1) / TPB, TPB, 0, stream>>>(emax, -__builtin_inff(), T * BN);
  fill_f32<<<(T * BN + TPB - 1) / TPB, TPB, 0, stream>>>(den, 0.f, T * BN);
  bias_rows<<<(nEC + TPB - 1) / TPB, TPB, 0, stream>>>(H1, g1_bias, ROWS);
  edge_score<<<(nE + TPB - 1) / TPB, TPB, 0, stream>>>(XL, XR, g1_att, edge_index, dyn, ebuf, emax);
  edge_num<<<(nE + TPB - 1) / TPB, TPB, 0, stream>>>(ebuf, emax, edge_index, dyn, numbuf, den);
  edge_agg<<<(nE * C + TPB - 1) / TPB, TPB, 0, stream>>>(numbuf, den, XL, edge_index, dyn, H1);
  relu_ip<<<(nEC + TPB - 1) / TPB, TPB, 0, stream>>>(H1, nEC);

  // 4. GATv2 layer 2
  gemm_wmma<<<g64, 32, 0, stream>>>(H1, g2_wl, g2_bl, nullptr, XL, ROWS, C, C, 0);
  gemm_wmma<<<g64, 32, 0, stream>>>(H1, g2_wr, g2_br, nullptr, XR, ROWS, C, C, 0);
  fill_f32<<<(T * BN + TPB - 1) / TPB, TPB, 0, stream>>>(emax, -__builtin_inff(), T * BN);
  fill_f32<<<(T * BN + TPB - 1) / TPB, TPB, 0, stream>>>(den, 0.f, T * BN);
  bias_rows<<<(nEC + TPB - 1) / TPB, TPB, 0, stream>>>(H2, g2_bias, ROWS);
  edge_score<<<(nE + TPB - 1) / TPB, TPB, 0, stream>>>(XL, XR, g2_att, edge_index, dyn, ebuf, emax);
  edge_num<<<(nE + TPB - 1) / TPB, TPB, 0, stream>>>(ebuf, emax, edge_index, dyn, numbuf, den);
  edge_agg<<<(nE * C + TPB - 1) / TPB, TPB, 0, stream>>>(numbuf, den, XL, edge_index, dyn, H2);

  // 5. temporal conv (over node axis, masked at t==0)
  tconvA<<<(nEC + TPB - 1) / TPB, TPB, 0, stream>>>(H2, tc_w1, tc_b1, Ct);
  tconvB<<<(nEC + TPB - 1) / TPB, TPB, 0, stream>>>(Ct, tc_w2, tc_b2, H2);

  // 6. transformer encoder layer
  remap_bt<<<(nEC + TPB - 1) / TPB, TPB, 0, stream>>>(H2, X2);
  gemm_wmma<<<dim3((ROWS / 16) * (192 / 16)), 32, 0, stream>>>(
      X2, attn_wqkv, attn_bqkv, nullptr, QKV, ROWS, C, 192, 0);
  attn_kernel<<<dim3(N, 4), 96, 0, stream>>>(QKV, Ob);
  gemm_wmma<<<g64, 32, 0, stream>>>(Ob, attn_wo, attn_bo, X2, Y1, ROWS, C, C, 0);
  ln_kernel<<<ROWS, 32, 0, stream>>>(Y1, ln1_g, ln1_b, XN);
  ffn_fused<<<ROWS / 16, 32, 0, stream>>>(XN, ff_w1, ff_b1, ff_w2, ff_b2, Y2);
  ln_kernel<<<ROWS, 32, 0, stream>>>(Y2, ln2_g, ln2_b, X3);

  // 7. head
  tmax_kernel<<<(B * N * C + TPB - 1) / TPB, TPB, 0, stream>>>(X3, tmx);
  combo_kernel<<<1, 128, 0, stream>>>(fus_w, fus_b, reg_w, reg_b, combo);
  final_kernel<<<(ROWS + TPB - 1) / TPB, TPB, 0, stream>>>(X3, tmx, combo, (float*)d_out);
}
